// NonPagedAttentionLayer_58832462021336
// MI455X (gfx1250) — compile-verified, matrix operand order checked by reading, not run on hardware
//
#include <hip/hip_runtime.h>
#include <hip/hip_bf16.h>

// ---------------- problem constants (from reference) ----------------
#define BB 2
#define SS 512
#define HH 32
#define DD 128
#define TT 2048
#define DM (HH * DD)          // 4096
#define TSTART (TT - SS)      // 1536

typedef __attribute__((ext_vector_type(16))) __bf16 v16bf;
typedef __attribute__((ext_vector_type(8)))  __bf16 v8bf;
typedef __attribute__((ext_vector_type(8)))  float  v8f;
typedef __attribute__((ext_vector_type(4)))  int    v4i;

// AS-qualified pointer types for the async-copy builtin (AS1 = global/__device__,
// AS3 = LDS/__shared__). Built from integers to sidestep addrspacecast
// restrictions: AS1 VA == generic VA; AS3 offset == low 32 bits of generic VA
// (ISA 10.2 aperture mapping).
typedef __attribute__((address_space(1))) v4i* gv4i_p;
typedef __attribute__((address_space(3))) v4i* lv4i_p;

__device__ __forceinline__ gv4i_p to_global_v4i(const void* p) {
  return (gv4i_p)(unsigned long long)p;
}
__device__ __forceinline__ lv4i_p to_lds_v4i(const void* p) {
  return (lv4i_p)(unsigned int)(unsigned long long)p;
}

// -------- feature detection (keeps compile safe on either toolchain) --------
#if defined(__has_builtin)
#  if __has_builtin(__builtin_amdgcn_global_load_async_to_lds_b128)
#    define USE_ASYNC_LDS 1
#  else
#    define USE_ASYNC_LDS 0
#  endif
#  if __has_builtin(__builtin_amdgcn_s_wait_asynccnt)
#    define HAVE_WAIT_ASYNC_BUILTIN 1
#  else
#    define HAVE_WAIT_ASYNC_BUILTIN 0
#  endif
#else
#  define USE_ASYNC_LDS 0
#  define HAVE_WAIT_ASYNC_BUILTIN 0
#endif

__device__ __forceinline__ void wait_async_0() {
#if HAVE_WAIT_ASYNC_BUILTIN
  __builtin_amdgcn_s_wait_asynccnt(0);
#else
  asm volatile("s_wait_asynccnt 0x0" ::: "memory");
#endif
}
__device__ __forceinline__ void wait_async_16() {
#if HAVE_WAIT_ASYNC_BUILTIN
  __builtin_amdgcn_s_wait_asynccnt(16);
#else
  asm volatile("s_wait_asynccnt 0x10" ::: "memory");
#endif
}

// ---------------- WMMA helpers ----------------
__device__ __forceinline__ v8f wmma_bf16(v16bf a, v16bf b, v8f c) {
  return __builtin_amdgcn_wmma_f32_16x16x32_bf16(
      /*neg_a=*/false, a, /*neg_b=*/false, b,
      /*c_mod=*/(short)0, c, /*reuse_a=*/false, /*reuse_b=*/false);
}

// A/B operand layout for 16-bit 16x16x32 WMMA (ISA 7.12.2):
// lane l holds row m = l&15; lanes 0-15 cover K in {0..7, 16..23},
// lanes 16-31 cover K in {8..15, 24..31}. row_ptr points at element [m, kbase].
__device__ __forceinline__ v16bf load_frag(const __bf16* row_ptr, int lane) {
  int khalf = (lane >> 4) << 3;                      // 0 or 8
  v8bf lo = *(const v8bf*)(row_ptr + khalf);         // K = khalf .. khalf+7
  v8bf hi = *(const v8bf*)(row_ptr + khalf + 16);    // K = 16+khalf .. 16+khalf+7
  v16bf f;
#pragma unroll
  for (int i = 0; i < 8; ++i) { f[i] = lo[i]; f[i + 8] = hi[i]; }
  return f;
}

// Same operand layout, K-dimension strided (used for V^T fragments out of LDS).
__device__ __forceinline__ v16bf load_frag_strided(const __bf16* base, int ld, int lane) {
  int khalf = (lane >> 4) << 3;
  v16bf f;
#pragma unroll
  for (int i = 0; i < 8; ++i) {
    f[i]     = base[(khalf + i) * ld];
    f[i + 8] = base[(16 + khalf + i) * ld];
  }
  return f;
}

// Single-wave LDS sync: lanes are lockstep, only need DS ops drained.
__device__ __forceinline__ void wave_lds_sync() {
  __builtin_amdgcn_wave_barrier();
  asm volatile("s_wait_dscnt 0x0" ::: "memory");
  __builtin_amdgcn_wave_barrier();
}

// ---------------- elementwise cast f32 -> bf16 ----------------
__global__ void cast_f32_bf16(const float* __restrict__ src, __bf16* __restrict__ dst, size_t n) {
  size_t i = (size_t)blockIdx.x * blockDim.x + threadIdx.x;
  size_t stride = (size_t)gridDim.x * blockDim.x;
  for (; i < n; i += stride) dst[i] = (__bf16)src[i];
}

// ---------------- generic GEMM: C[M,N] = A[M,K] * B[N,K]^T ----------------
// Block = 256 threads = 8 waves. Wave w owns a 16x64 output tile:
// m0 = blockIdx.y*16, n0 = (blockIdx.x*8 + w)*64. A fragment reused across 4 B
// tiles. K-loop unrolled by 2 with ping-pong fragment sets: loads for the idle
// set issue while WMMAs consume the live set -- no register rotation copies.
// Requires K % 64 == 0 and K >= 128 (true here: K = 4096).
__global__ __launch_bounds__(256) void gemm_bf16(const __bf16* __restrict__ A,
                                                 const __bf16* __restrict__ Bw,
                                                 float* __restrict__ C,
                                                 int M, int N, int K) {
  int lane = threadIdx.x & 31;
  int wave = threadIdx.x >> 5;
  int m0 = blockIdx.y * 16;
  int n0 = (blockIdx.x * 8 + wave) * 64;
  const __bf16* arow = A + (size_t)(m0 + (lane & 15)) * K;
  const __bf16* brow[4];
#pragma unroll
  for (int j = 0; j < 4; ++j)
    brow[j] = Bw + (size_t)(n0 + j * 16 + (lane & 15)) * K;

  v8f acc[4];
#pragma unroll
  for (int j = 0; j < 4; ++j) acc[j] = (v8f)(0.0f);

  v16bf a0, a1, b0[4], b1[4];
  a0 = load_frag(arow, lane);
#pragma unroll
  for (int j = 0; j < 4; ++j) b0[j] = load_frag(brow[j], lane);

  int k = 0;
  for (; k + 64 < K; k += 64) {
    __builtin_prefetch(arow + k + 256, 0, 1);   // -> global_prefetch_b8
    // phase 0: load set1 (k+32), compute on set0 (k)
    a1 = load_frag(arow + k + 32, lane);
#pragma unroll
    for (int j = 0; j < 4; ++j) b1[j] = load_frag(brow[j] + k + 32, lane);
#pragma unroll
    for (int j = 0; j < 4; ++j) acc[j] = wmma_bf16(a0, b0[j], acc[j]);
    // phase 1: load set0 (k+64), compute on set1 (k+32)
    a0 = load_frag(arow + k + 64, lane);
#pragma unroll
    for (int j = 0; j < 4; ++j) b0[j] = load_frag(brow[j] + k + 64, lane);
#pragma unroll
    for (int j = 0; j < 4; ++j) acc[j] = wmma_bf16(a1, b1[j], acc[j]);
  }
  // tail: set0 holds fragments for k == K-64; one more pair remains
  a1 = load_frag(arow + k + 32, lane);
#pragma unroll
  for (int j = 0; j < 4; ++j) b1[j] = load_frag(brow[j] + k + 32, lane);
#pragma unroll
  for (int j = 0; j < 4; ++j) acc[j] = wmma_bf16(a0, b0[j], acc[j]);
#pragma unroll
  for (int j = 0; j < 4; ++j) acc[j] = wmma_bf16(a1, b1[j], acc[j]);

  int rbase = (lane >> 4) << 3;
#pragma unroll
  for (int j = 0; j < 4; ++j)
#pragma unroll
    for (int r = 0; r < 8; ++r)
      C[(size_t)(m0 + rbase + r) * N + n0 + j * 16 + (lane & 15)] = acc[j][r];
}

// ---------------- RoPE + cache scatter ----------------
// One thread per (b, s, h, pair i in 0..63). qkv is fp32 [B*S, 3*DM].
__global__ void rope_scatter(const float* __restrict__ qkv,
                             const int* __restrict__ batch_idx,
                             const int* __restrict__ input_pos,
                             __bf16* __restrict__ qb,      // [B,H,S,D]
                             __bf16* __restrict__ kall,    // [B,H,T,D]
                             __bf16* __restrict__ vall) {  // [B,H,T,D]
  size_t idx = (size_t)blockIdx.x * blockDim.x + threadIdx.x;  // < B*S*H*64
  int i = idx & 63;
  int h = (idx >> 6) & (HH - 1);
  int s = (idx >> 11) & (SS - 1);
  int b = (int)(idx >> 20);

  int pos = input_pos[b * SS + s];
  int bi  = batch_idx[b];
  int d0  = 2 * i;
  float inv = __powf(10000.0f, -(float)d0 / (float)DD);
  float ang = (float)pos * inv;
  float c = __cosf(ang), sn = __sinf(ang);

  size_t base = ((size_t)b * SS + s) * (3 * DM);
  int dmq = h * DD + d0;
  float q0 = qkv[base + dmq],            q1 = qkv[base + dmq + 1];
  float k0 = qkv[base + DM + dmq],       k1 = qkv[base + DM + dmq + 1];
  float v0 = qkv[base + 2 * DM + dmq],   v1 = qkv[base + 2 * DM + dmq + 1];

  size_t qo = (((size_t)b * HH + h) * SS + s) * DD + d0;
  qb[qo]     = (__bf16)(q0 * c - q1 * sn);
  qb[qo + 1] = (__bf16)(q1 * c + q0 * sn);

  size_t ko = (((size_t)bi * HH + h) * TT + pos) * DD + d0;
  kall[ko]     = (__bf16)(k0 * c - k1 * sn);
  kall[ko + 1] = (__bf16)(k1 * c + k0 * sn);
  vall[ko]     = (__bf16)v0;
  vall[ko + 1] = (__bf16)v1;
}

// ---------------- flash attention (online softmax, bf16 WMMA) ----------------
// Block = 128 threads = 4 waves; each wave owns one 16-row query tile.
// V tiles (32 keys x 128 dims, 8 KB) are staged into LDS with async copies,
// double-buffered across key tiles; PV fragments then read from LDS.
// grid = (S/16/4, H, B).
__global__ __launch_bounds__(128) void attn_kernel(const __bf16* __restrict__ qb,
                                                   const __bf16* __restrict__ kall,
                                                   const __bf16* __restrict__ vall,
                                                   __bf16* __restrict__ yb) {  // [B*S, DM]
  __shared__ __align__(16) __bf16 lp[4][16 * 32];         // per-wave P staging
  __shared__ __align__(16) __bf16 vstage[4][2][32 * 128]; // per-wave V double buffer

  int lane = threadIdx.x & 31;
  int wave = threadIdx.x >> 5;
  int col  = lane & 15;
  int h = blockIdx.y, b = blockIdx.z;
  int qtile = blockIdx.x * 4 + wave;

  // Q fragments (A operand), preloaded: D=128 -> 4 K-steps of 32.
  const __bf16* qrow = qb + (((size_t)b * HH + h) * SS + qtile * 16 + col) * DD;
  v16bf qf[4];
#pragma unroll
  for (int kk = 0; kk < 4; ++kk) qf[kk] = load_frag(qrow + kk * 32, lane);

  const __bf16* kbase = kall + ((size_t)b * HH + h) * TT * DD;
  const __bf16* vbase = vall + ((size_t)b * HH + h) * TT * DD;

  // async stage of one 32x128 V tile: 16 issues x 32 lanes x 16B = 8 KB
  auto prefetch_v = [&](int t0, int buf) {
    const __bf16* gsrc = vbase + (size_t)t0 * DD;
    __bf16* ldst = &vstage[wave][buf][0];
#if USE_ASYNC_LDS
#pragma unroll
    for (int i = 0; i < 16; ++i)
      __builtin_amdgcn_global_load_async_to_lds_b128(
          to_global_v4i(gsrc + i * 256 + lane * 8),
          to_lds_v4i(ldst + i * 256 + lane * 8), 0, 0);
#else
#pragma unroll
    for (int i = 0; i < 16; ++i)
      *(v8bf*)(ldst + i * 256 + lane * 8) = *(const v8bf*)(gsrc + i * 256 + lane * 8);
#endif
  };

  v8f o[8];
#pragma unroll
  for (int j = 0; j < 8; ++j) o[j] = (v8f)(0.0f);
  float mi[8], li[8];
#pragma unroll
  for (int r = 0; r < 8; ++r) { mi[r] = -3.0e38f; li[r] = 0.0f; }

  int rowbase = (lane >> 4) << 3;
  const float scale = 0.08838834764831845f;  // 1/sqrt(128)
  int tmax = TSTART + qtile * 16 + 16;       // causal bound for this tile

  prefetch_v(0, 0);

  for (int t0 = 0; t0 < tmax; t0 += 32) {
    int buf = (t0 >> 5) & 1;
    bool have_next = (t0 + 32) < tmax;
    if (have_next) prefetch_v(t0 + 32, buf ^ 1);

    // ---- scores: two 16x16 tiles over 32 keys (K from global/L2) ----
    v8f st[2];
#pragma unroll
    for (int half = 0; half < 2; ++half) {
      st[half] = (v8f)(0.0f);
      const __bf16* krow = kbase + (size_t)(t0 + half * 16 + col) * DD;
#pragma unroll
      for (int kk = 0; kk < 4; ++kk)
        st[half] = wmma_bf16(qf[kk], load_frag(krow + kk * 32, lane), st[half]);
    }

    // ---- online softmax stats (elem r == row rowbase+r; lane group == cols) ----
    float f[8];
#pragma unroll
    for (int r = 0; r < 8; ++r) {
      int qpos = TSTART + qtile * 16 + rowbase + r;
      float s0 = st[0][r] * scale; if (t0 + col > qpos)      s0 = -3.0e38f;
      float s1 = st[1][r] * scale; if (t0 + 16 + col > qpos) s1 = -3.0e38f;
      float rm = fmaxf(s0, s1);
#pragma unroll
      for (int m = 1; m < 16; m <<= 1) rm = fmaxf(rm, __shfl_xor(rm, m, 32));
      float nm = fmaxf(mi[r], rm);
      f[r] = __expf(mi[r] - nm);
      float p0 = __expf(s0 - nm);
      float p1 = __expf(s1 - nm);
      lp[wave][(rowbase + r) * 32 + col]      = (__bf16)p0;
      lp[wave][(rowbase + r) * 32 + 16 + col] = (__bf16)p1;
      float prs = p0 + p1;
#pragma unroll
      for (int m = 1; m < 16; m <<= 1) prs += __shfl_xor(prs, m, 32);
      li[r] = li[r] * f[r] + prs;
      mi[r] = nm;
    }

    wave_lds_sync();  // P staged in LDS (C layout) -> reload as A layout
    v16bf pf = load_frag(&lp[wave][(size_t)col * 32], lane);

    // current V tile must be resident; next tile may stay in flight
    if (have_next) wait_async_16(); else wait_async_0();

    // ---- PV: O(16x128) += P(16x32) * V(32x128), V^T fragments from LDS ----
    const __bf16* vt = &vstage[wave][buf][0];
#pragma unroll
    for (int j = 0; j < 8; ++j) {
      v8f oj = o[j];
#pragma unroll
      for (int r = 0; r < 8; ++r) oj[r] *= f[r];
      v16bf vf = load_frag_strided(vt + j * 16 + col, DD, lane);
      o[j] = wmma_bf16(pf, vf, oj);
    }
    wave_lds_sync();  // safe to overwrite lp next iteration
  }

  // ---- epilogue: normalize, store y as bf16 [B*S, DM] (dm = h*D + d) ----
#pragma unroll
  for (int j = 0; j < 8; ++j) {
#pragma unroll
    for (int r = 0; r < 8; ++r) {
      int srow = qtile * 16 + rowbase + r;
      yb[((size_t)b * SS + srow) * DM + h * DD + j * 16 + col] =
          (__bf16)(o[j][r] / li[r]);
    }
  }
}

// ---------------- host side ----------------
extern "C" void kernel_launch(void* const* d_in, const int* in_sizes, int n_in,
                              void* d_out, int out_size, void* d_ws, size_t ws_size,
                              hipStream_t stream) {
  const float* x       = (const float*)d_in[0];   // [B,S,DM]
  const float* wqkv    = (const float*)d_in[1];   // [3*DM, DM]
  const float* wo      = (const float*)d_in[2];   // [DM, DM]
  const float* k_cache = (const float*)d_in[3];   // [B,H,T,D]
  const float* v_cache = (const float*)d_in[4];   // [B,H,T,D]
  const int*   batch_idx = (const int*)d_in[5];
  const int*   input_pos = (const int*)d_in[6];
  float* out = (float*)d_out;                     // [B,S,DM] fp32

  // workspace carve-up (256B aligned)
  char* ws = (char*)d_ws;
  size_t off = 0;
  auto carve = [&](size_t bytes) {
    char* p = ws + off;
    off = (off + bytes + 255) & ~(size_t)255;
    return p;
  };
  const size_t nx    = (size_t)BB * SS * DM;      // 4M
  const size_t nwqkv = (size_t)3 * DM * DM;       // 50.3M
  const size_t nwo   = (size_t)DM * DM;           // 16.8M
  const size_t ncache= (size_t)BB * HH * TT * DD; // 16.8M

  __bf16* x_bf    = (__bf16*)carve(nx * 2);
  __bf16* wqkv_bf = (__bf16*)carve(nwqkv * 2);
  __bf16* wo_bf   = (__bf16*)carve(nwo * 2);
  float*  qkv_f32 = (float*) carve((size_t)BB * SS * 3 * DM * 4);
  __bf16* q_bf    = (__bf16*)carve(nx * 2);
  __bf16* k_all   = (__bf16*)carve(ncache * 2);
  __bf16* v_all   = (__bf16*)carve(ncache * 2);
  __bf16* y_bf    = (__bf16*)carve(nx * 2);
  (void)ws_size;

  // 1) casts (grid-stride)
  cast_f32_bf16<<<2048, 256, 0, stream>>>(x,       x_bf,    nx);
  cast_f32_bf16<<<4096, 256, 0, stream>>>(wqkv,    wqkv_bf, nwqkv);
  cast_f32_bf16<<<4096, 256, 0, stream>>>(wo,      wo_bf,   nwo);
  cast_f32_bf16<<<4096, 256, 0, stream>>>(k_cache, k_all,   ncache);
  cast_f32_bf16<<<4096, 256, 0, stream>>>(v_cache, v_all,   ncache);

  // 2) QKV projection: M=1024, N=12288, K=4096
  {
    dim3 grid(3 * DM / 512, BB * SS / 16);
    gemm_bf16<<<grid, 256, 0, stream>>>(x_bf, wqkv_bf, qkv_f32,
                                        BB * SS, 3 * DM, DM);
  }

  // 3) RoPE + scatter new K/V into bf16 cache copies
  {
    size_t threads = (size_t)BB * SS * HH * 64;   // 2^21
    rope_scatter<<<(unsigned)(threads / 256), 256, 0, stream>>>(
        qkv_f32, batch_idx, input_pos, q_bf, k_all, v_all);
  }

  // 4) attention
  {
    dim3 grid(SS / 16 / 4, HH, BB);
    attn_kernel<<<grid, 128, 0, stream>>>(q_bf, k_all, v_all, y_bf);
  }

  // 5) output projection: M=1024, N=4096, K=4096 -> fp32 out
  {
    dim3 grid(DM / 512, BB * SS / 16);
    gemm_bf16<<<grid, 256, 0, stream>>>(y_bf, wo_bf, out,
                                        BB * SS, DM, DM);
  }
  (void)in_sizes; (void)n_in; (void)out_size;
}
